// LuongSeq2Seq_15444702397224
// MI455X (gfx1250) — matrix operand order repaired and checked by reading, not compiled, not run on hardware
//
#include <hip/hip_runtime.h>

// ---------------- problem constants ----------------
#define B_   64
#define S_   128
#define T_   128
#define E_   512
#define H_   1024
#define G3H  3072      // 3*H
#define V_   32000

// ---------------- vector types for WMMA ----------------
typedef __attribute__((ext_vector_type(16))) __bf16 v16bf;
typedef __attribute__((ext_vector_type(8)))  float  v8f;
typedef __attribute__((ext_vector_type(16))) short  v16s;
typedef __attribute__((ext_vector_type(8)))  short  v8s;

__device__ __forceinline__ unsigned short f2bf(float f) {
  union { float f; unsigned u; } v; v.f = f;
  unsigned r = v.u + 0x7FFFu + ((v.u >> 16) & 1u);   // round-to-nearest-even
  return (unsigned short)(r >> 16);
}

// =====================================================================
// Generic bf16 GEMM:  C[M,N] = A[M,K] * W[N,K]^T + bias[N]   (f32 accum)
// One wave computes a 16x64 tile via v_wmma_f32_16x16x32_bf16.
// Block = 128 threads (4 waves stacked along M); grid = (N/64, M/64).
// Fragments use contiguous k order (k = 16*half + s), not the ISA table order (wave32).
// =====================================================================
__global__ void gemm_bf16_wmma(const unsigned short* __restrict__ A,
                               const unsigned short* __restrict__ W,
                               const float* __restrict__ bias,
                               float* __restrict__ C,
                               int K, int N)
{
  const int lane = threadIdx.x & 31;
  const int wave = threadIdx.x >> 5;
  const int half = lane >> 4;      // 0: lanes 0-15, 1: lanes 16-31
  const int l16  = lane & 15;
  const int m0 = blockIdx.y * 64 + wave * 16;
  const int n0 = blockIdx.x * 64;

  v8f acc[4] = {};

  const unsigned short* arow  = A + (size_t)(m0 + l16) * K;
  const unsigned short* wbase = W + (size_t)(n0 + l16) * K + 16 * half;

  for (int k0 = 0; k0 < K; k0 += 32) {
    // A 16x32 fragment: elems 0..15 = K[k0+16h .. +15]
    v8s alo = *(const v8s*)(arow + k0 + 16 * half);
    v8s ahi = *(const v8s*)(arow + k0 + 16 * half + 8);
    v16s av = __builtin_shufflevector(alo, ahi,
                0,1,2,3,4,5,6,7,8,9,10,11,12,13,14,15);
    v16bf a = __builtin_bit_cast(v16bf, av);
#pragma unroll
    for (int nt = 0; nt < 4; ++nt) {
      // B 32x16 fragment: lane holds weight row (n0+nt*16+l16), 16 contiguous K
      v16s bv = *(const v16s*)(wbase + (size_t)nt * 16 * K + k0);
      v16bf b = __builtin_bit_cast(v16bf, bv);
      acc[nt] = __builtin_amdgcn_wmma_f32_16x16x32_bf16(
          false, a, false, b, (short)0, acc[nt], false, false);
    }
  }

#pragma unroll
  for (int nt = 0; nt < 4; ++nt) {
    const int col = n0 + nt * 16 + l16;
    const float bb = bias[col];
#pragma unroll
    for (int r = 0; r < 8; ++r) {
      C[(size_t)(m0 + r + 8 * half) * N + col] = acc[nt][r] + bb;
    }
  }
}

// ---------------- fp32 -> bf16 cast ----------------
__global__ void cast_bf16_k(const float* __restrict__ src,
                            unsigned short* __restrict__ dst, long n)
{
  long i = (long)blockIdx.x * blockDim.x + threadIdx.x;
  if (i < n) dst[i] = f2bf(src[i]);
}

// ---------------- embedding gather + cast ----------------
__global__ void embed_bf16_k(const int* __restrict__ tok,
                             const float* __restrict__ emb,
                             unsigned short* __restrict__ out, long total)
{
  long idx = (long)blockIdx.x * blockDim.x + threadIdx.x;
  if (idx >= total) return;
  int  e = (int)(idx % E_);
  long r = idx / E_;
  out[idx] = f2bf(emb[(size_t)tok[r] * E_ + e]);
}

// ---------------- hidden-state init ----------------
__global__ void init_h_k(float* __restrict__ h, unsigned short* __restrict__ hbf,
                         const float* __restrict__ src, int n)
{
  int i = blockIdx.x * blockDim.x + threadIdx.x;
  if (i >= n) return;
  float v = src ? src[i] : 0.0f;
  h[i] = v;
  hbf[i] = f2bf(v);
}

// ---------------- fused GRU gate / state update ----------------
__global__ void gru_gate_k(const float* __restrict__ xw,   // [B,T,3H]
                           const float* __restrict__ hw,   // [B,3H]
                           float* __restrict__ h,          // [B,H]
                           unsigned short* __restrict__ hbf,
                           unsigned short* __restrict__ outbf, // [B,T,H] bf16
                           float* __restrict__ ctx,            // [B,T,H] f32 or null
                           const int* __restrict__ len,        // null => all valid
                           int t)
{
  int idx = blockIdx.x * blockDim.x + threadIdx.x;   // over B*H
  int b = idx >> 10;                                  // /H_
  int j = idx & (H_ - 1);
  size_t xrow = ((size_t)b * T_ + t) * G3H;
  size_t hrow = (size_t)b * G3H;
  float xr = xw[xrow + j], xz = xw[xrow + H_ + j], xn = xw[xrow + 2 * H_ + j];
  float hr = hw[hrow + j], hz = hw[hrow + H_ + j], hn = hw[hrow + 2 * H_ + j];
  float r = 1.0f / (1.0f + __expf(-(xr + hr)));
  float z = 1.0f / (1.0f + __expf(-(xz + hz)));
  float n = tanhf(xn + r * hn);
  float hp = h[idx];
  float hnew = (1.0f - z) * n + z * hp;
  bool m = len ? (t < len[b]) : true;
  hnew = m ? hnew : hp;                 // freeze past sequence end
  h[idx] = hnew;
  hbf[idx] = f2bf(hnew);
  float o = m ? hnew : 0.0f;            // padded outputs are zero
  size_t orow = ((size_t)b * T_ + t) * H_ + j;
  outbf[orow] = f2bf(o);
  if (ctx) ctx[orow] = o;
}

// ---------------- attention: masked dot scores + softmax ----------------
__global__ void attn_score_k(const float* __restrict__ q,    // [B,H]
                             const float* __restrict__ ctx,  // [B,S,H]
                             const int* __restrict__ enc_len,
                             float* __restrict__ a)          // [B,S]
{
  int b = blockIdx.x;
  int s = threadIdx.x;                 // 128 threads = S
  const float* qb = q + (size_t)b * H_;
  const float* cb = ctx + ((size_t)b * S_ + s) * H_;
  float e = 0.0f;
  for (int k = 0; k < H_; ++k) e += qb[k] * cb[k];
  e = (s < enc_len[b]) ? e : -1e10f;

  __shared__ float red[S_];
  red[s] = e; __syncthreads();
  for (int off = S_ / 2; off > 0; off >>= 1) {
    if (s < off) red[s] = fmaxf(red[s], red[s + off]);
    __syncthreads();
  }
  float mx = red[0]; __syncthreads();
  float ex = __expf(e - mx);
  red[s] = ex; __syncthreads();
  for (int off = S_ / 2; off > 0; off >>= 1) {
    if (s < off) red[s] += red[s + off];
    __syncthreads();
  }
  a[(size_t)b * S_ + s] = ex / red[0];
}

// ---------------- attention: context vector ----------------
__global__ void attn_ctx_k(const float* __restrict__ a,     // [B,S]
                           const float* __restrict__ ctx,   // [B,S,H]
                           float* __restrict__ c)           // [B,H]
{
  int b = blockIdx.x;
  for (int j = threadIdx.x; j < H_; j += blockDim.x) {
    float acc = 0.0f;
    for (int s = 0; s < S_; ++s)
      acc += a[(size_t)b * S_ + s] * ctx[((size_t)b * S_ + s) * H_ + j];
    c[(size_t)b * H_ + j] = acc;
  }
}

// ---------------- concat [q, c] -> bf16 ----------------
__global__ void cat_bf16_k(const float* __restrict__ q, const float* __restrict__ c,
                           unsigned short* __restrict__ cat)
{
  int idx = blockIdx.x * blockDim.x + threadIdx.x;   // B*2H
  int b = idx >> 11;
  int j = idx & (2 * H_ - 1);
  float v = (j < H_) ? q[(size_t)b * H_ + j] : c[(size_t)b * H_ + (j - H_)];
  cat[idx] = f2bf(v);
}

// =====================================================================
// Host orchestration
// =====================================================================
static void run_gru_layer(const unsigned short* Xbf, int Kin,
                          const unsigned short* Wih, const float* bih,
                          const unsigned short* Whh, const float* bhh,
                          const float* h0, const int* len,
                          unsigned short* outbf, float* ctx,
                          float* xw, float* hw, float* h, unsigned short* hbf,
                          hipStream_t stream)
{
  // input projection for all timesteps at once (parallel big GEMM)
  gemm_bf16_wmma<<<dim3(G3H / 64, (B_ * T_) / 64), 128, 0, stream>>>(
      Xbf, Wih, bih, xw, Kin, G3H);
  init_h_k<<<(B_ * H_) / 256, 256, 0, stream>>>(h, hbf, h0, B_ * H_);
  for (int t = 0; t < T_; ++t) {
    gemm_bf16_wmma<<<dim3(G3H / 64, 1), 128, 0, stream>>>(
        hbf, Whh, bhh, hw, H_, G3H);
    gru_gate_k<<<(B_ * H_) / 256, 256, 0, stream>>>(
        xw, hw, h, hbf, outbf, ctx, len, t);
  }
}

extern "C" void kernel_launch(void* const* d_in, const int* in_sizes, int n_in,
                              void* d_out, int out_size, void* d_ws, size_t ws_size,
                              hipStream_t stream) {
  (void)in_sizes; (void)n_in; (void)out_size; (void)ws_size;

  const int*   enc_x    = (const int*)d_in[0];
  const int*   enc_len  = (const int*)d_in[1];
  const int*   dec_x    = (const int*)d_in[2];
  const float* emb_enc  = (const float*)d_in[5];
  const float* encWih0  = (const float*)d_in[6];
  const float* encWhh0  = (const float*)d_in[7];
  const float* enc_bih0 = (const float*)d_in[8];
  const float* enc_bhh0 = (const float*)d_in[9];
  const float* encWih1  = (const float*)d_in[10];
  const float* encWhh1  = (const float*)d_in[11];
  const float* enc_bih1 = (const float*)d_in[12];
  const float* enc_bhh1 = (const float*)d_in[13];
  const float* emb_dec  = (const float*)d_in[14];
  const float* decWih0  = (const float*)d_in[15];
  const float* decWhh0  = (const float*)d_in[16];
  const float* dec_bih0 = (const float*)d_in[17];
  const float* dec_bhh0 = (const float*)d_in[18];
  const float* decWih1  = (const float*)d_in[19];
  const float* decWhh1  = (const float*)d_in[20];
  const float* dec_bih1 = (const float*)d_in[21];
  const float* dec_bhh1 = (const float*)d_in[22];
  const float* out_W    = (const float*)d_in[23];
  const float* out_b    = (const float*)d_in[24];

  // ---------------- workspace partition ----------------
  char* ws = (char*)d_ws;
  size_t off = 0;
  auto take = [&](size_t bytes) -> char* {
    off = (off + 255) & ~(size_t)255;
    char* p = ws + off; off += bytes; return p;
  };

  unsigned short* wEncWih0 = (unsigned short*)take(2ull * G3H * E_);
  unsigned short* wEncWhh0 = (unsigned short*)take(2ull * G3H * H_);
  unsigned short* wEncWih1 = (unsigned short*)take(2ull * G3H * H_);
  unsigned short* wEncWhh1 = (unsigned short*)take(2ull * G3H * H_);
  unsigned short* wDecWih0 = (unsigned short*)take(2ull * G3H * E_);
  unsigned short* wDecWhh0 = (unsigned short*)take(2ull * G3H * H_);
  unsigned short* wDecWih1 = (unsigned short*)take(2ull * G3H * H_);
  unsigned short* wDecWhh1 = (unsigned short*)take(2ull * G3H * H_);
  unsigned short* wOut     = (unsigned short*)take(2ull * V_ * 2 * H_);
  unsigned short* embBf    = (unsigned short*)take(2ull * B_ * T_ * E_);
  unsigned short* out0Bf   = (unsigned short*)take(2ull * B_ * T_ * H_);
  unsigned short* out1Bf   = (unsigned short*)take(2ull * B_ * T_ * H_);
  float*          xw       = (float*)take(4ull * B_ * T_ * G3H);
  float*          ctx      = (float*)take(4ull * B_ * S_ * H_);
  float*          hw       = (float*)take(4ull * B_ * G3H);
  float*          h        = (float*)take(4ull * B_ * H_);
  unsigned short* hbf      = (unsigned short*)take(2ull * B_ * H_);
  float*          henc0    = (float*)take(4ull * B_ * H_);
  float*          henc1    = (float*)take(4ull * B_ * H_);
  float*          attA     = (float*)take(4ull * B_ * S_);
  float*          cvec     = (float*)take(4ull * B_ * H_);
  unsigned short* catBf    = (unsigned short*)take(2ull * B_ * 2 * H_);

  // ---------------- cast weights to bf16 (resident in 192MB L2 afterwards) ----
  auto cast = [&](const float* s, unsigned short* d, size_t n) {
    cast_bf16_k<<<(unsigned)((n + 255) / 256), 256, 0, stream>>>(s, d, (long)n);
  };
  cast(encWih0, wEncWih0, (size_t)G3H * E_);
  cast(encWhh0, wEncWhh0, (size_t)G3H * H_);
  cast(encWih1, wEncWih1, (size_t)G3H * H_);
  cast(encWhh1, wEncWhh1, (size_t)G3H * H_);
  cast(decWih0, wDecWih0, (size_t)G3H * E_);
  cast(decWhh0, wDecWhh0, (size_t)G3H * H_);
  cast(decWih1, wDecWih1, (size_t)G3H * H_);
  cast(decWhh1, wDecWhh1, (size_t)G3H * H_);
  cast(out_W,   wOut,     (size_t)V_ * 2 * H_);

  // ---------------- encoder ----------------
  embed_bf16_k<<<(B_ * S_ * E_) / 256, 256, 0, stream>>>(
      enc_x, emb_enc, embBf, (long)B_ * S_ * E_);
  run_gru_layer(embBf, E_, wEncWih0, enc_bih0, wEncWhh0, enc_bhh0,
                nullptr, enc_len, out0Bf, nullptr, xw, hw, h, hbf, stream);
  hipMemcpyAsync(henc0, h, 4ull * B_ * H_, hipMemcpyDeviceToDevice, stream);
  run_gru_layer(out0Bf, H_, wEncWih1, enc_bih1, wEncWhh1, enc_bhh1,
                nullptr, enc_len, out1Bf, ctx, xw, hw, h, hbf, stream);
  hipMemcpyAsync(henc1, h, 4ull * B_ * H_, hipMemcpyDeviceToDevice, stream);

  // ---------------- decoder (teacher-forced, mask = all valid) ----------------
  embed_bf16_k<<<(B_ * T_ * E_) / 256, 256, 0, stream>>>(
      dec_x, emb_dec, embBf, (long)B_ * T_ * E_);
  run_gru_layer(embBf, E_, wDecWih0, dec_bih0, wDecWhh0, dec_bhh0,
                henc0, nullptr, out0Bf, nullptr, xw, hw, h, hbf, stream);
  run_gru_layer(out0Bf, H_, wDecWih1, dec_bih1, wDecWhh1, dec_bhh1,
                henc1, nullptr, out1Bf, nullptr, xw, hw, h, hbf, stream);
  // h now holds q = final top-layer decoder hidden [B,H]

  // ---------------- Luong attention ----------------
  attn_score_k<<<B_, S_, 0, stream>>>(h, ctx, enc_len, attA);
  attn_ctx_k<<<B_, 256, 0, stream>>>(attA, ctx, cvec);
  cat_bf16_k<<<(B_ * 2 * H_) / 256, 256, 0, stream>>>(h, cvec, catBf);

  // ---------------- output projection [64,2048] x [32000,2048]^T ----------------
  gemm_bf16_wmma<<<dim3(V_ / 64, 1), 128, 0, stream>>>(
      catBf, wOut, out_b, (float*)d_out, 2 * H_, V_);
}
